// modelGcn_20650202759858
// MI455X (gfx1250) — matrix operand order repaired and checked by reading, not compiled, not run on hardware
//
#include <hip/hip_runtime.h>
#include <hip/hip_bf16.h>

typedef __attribute__((ext_vector_type(16))) _Float16 v16h;
typedef __attribute__((ext_vector_type(8)))  float    v8f;

// ---------------------------------------------------------------------------
// GEMM: Out[nrows, NCOL] = X[nrows, 64] @ W[64, NCOL]   (f16 WMMA, f32 accum)
// One wave32 per 16-row tile. K=64 -> two 16x16x32 WMMA steps per col-tile.
//
// W is staged into LDS pre-packed in WMMA B-fragment order:
//   fragment (ks, t), lane L, halves j=0..15  <-  W[(ks*32 + 8*(L>>4) + (j&7) + 16*(j>>3))*NCOL
//                                                  + t*16 + (L&15)]
// stored contiguously at wl[(((ks*(NCOL/16)) + t)*32 + L)*16 + j], so each
// lane's B fragment is ONE aligned 32-byte LDS load (ds_load_b128 x2).
//
// A layout (16-bit 16x32): lane L (L&15 = row M); halves 0..7 -> K = 8*(L>>4)+j,
//                          halves 8..15 -> K = 16 + 8*(L>>4) + (j-8).
// C/D layout (f32 16x16):  VGPR r, lane L -> M = 8*(L>>4)+r, N = L&15.
// ---------------------------------------------------------------------------
template <int NCOL>
__global__ void gcn_gemm_wmma(const float* __restrict__ X,
                              const float* __restrict__ W,
                              float* __restrict__ Out,
                              int ntiles) {
    constexpr int CT = NCOL / 16;                 // col tiles
    __shared__ __align__(16) _Float16 wl[64 * NCOL];

    // Stage W -> LDS in fragment-packed order.
    for (int idx = threadIdx.x; idx < 64 * NCOL; idx += blockDim.x) {
        int j    = idx & 15;
        int rest = idx >> 4;
        int L    = rest & 31; rest >>= 5;
        int t    = rest % CT;
        int ks   = rest / CT;
        int k    = ks * 32 + 8 * (L >> 4) + (j & 7) + 16 * (j >> 3);
        int col  = t * 16 + (L & 15);
        wl[idx]  = (_Float16)W[k * NCOL + col];
    }
    __syncthreads();

    const int wave = blockIdx.x * (blockDim.x >> 5) + (threadIdx.x >> 5);
    if (wave >= ntiles) return;                   // wave-uniform exit (EXEC full)
    const int lane = threadIdx.x & 31;
    const int row  = lane & 15;
    const int hi   = lane >> 4;                   // 0 or 1
    const int m0   = wave * 16;
    const float* xrow = X + (size_t)(m0 + row) * 64;

    v8f acc[CT];
#pragma unroll
    for (int t = 0; t < CT; ++t) acc[t] = (v8f){0.f,0.f,0.f,0.f,0.f,0.f,0.f,0.f};

#pragma unroll
    for (int ks = 0; ks < 2; ++ks) {
        const int k0 = ks * 32;
        // --- A fragment (vectorized global loads + cvt) ---
        v16h a;
        const int koff = hi * 8;
#pragma unroll
        for (int j = 0; j < 8; ++j) a[j]     = (_Float16)xrow[k0 + koff + j];
#pragma unroll
        for (int j = 0; j < 8; ++j) a[8 + j] = (_Float16)xrow[k0 + 16 + koff + j];

        // --- B fragments: one aligned 32B LDS load each ---
#pragma unroll
        for (int t = 0; t < CT; ++t) {
            const v16h b = *(const v16h*)&wl[((ks * CT + t) * 32 + lane) * 16];
            acc[t] = __builtin_amdgcn_wmma_f32_16x16x32_f16(
                false, a, false, b, (short)0, acc[t], false, false);
        }
    }

#pragma unroll
    for (int t = 0; t < CT; ++t) {
        const int col = t * 16 + (lane & 15);
#pragma unroll
        for (int r = 0; r < 8; ++r) {
            const int m = m0 + hi * 8 + r;
            Out[(size_t)m * NCOL + col] = acc[t][r];
        }
    }
}

// ---------------------------------------------------------------------------
// In-degree (dst side) via float atomics; self-loop folded in later (+1).
// ---------------------------------------------------------------------------
__global__ void gcn_degree(const int* __restrict__ dst, float* __restrict__ deg, int e) {
    int i = blockIdx.x * blockDim.x + threadIdx.x;
    if (i < e) atomicAdd(&deg[dst[i]], 1.0f);
}

__global__ void gcn_dinv(const float* __restrict__ deg, float* __restrict__ dinv, int n) {
    int i = blockIdx.x * blockDim.x + threadIdx.x;
    if (i < n) dinv[i] = rsqrtf(deg[i] + 1.0f);   // +1 = self loop; always >= 1
}

// ---------------------------------------------------------------------------
// Edge scatter: acc[dst] += T[src] * dinv[src]*dinv[dst].  Thread = (edge,feat).
// Each wave covers (a slice of) one edge's features -> edge loads are
// wave-uniform (scalarized); T gather is coalesced.
// ---------------------------------------------------------------------------
template <int F>
__global__ void gcn_scatter(const float* __restrict__ T,
                            const int* __restrict__ src,
                            const int* __restrict__ dst,
                            const float* __restrict__ dinv,
                            float* __restrict__ acc, int e) {
    int idx = blockIdx.x * blockDim.x + threadIdx.x;
    int ed = idx / F;
    int f  = idx - ed * F;
    if (ed >= e) return;
    int s = src[ed], d = dst[ed];
    float w = dinv[s] * dinv[d];
    atomicAdd(&acc[(size_t)d * F + f], T[(size_t)s * F + f] * w);
}

// ---------------------------------------------------------------------------
// Epilogue: out = relu(acc + T*dinv^2 + b)   (self-loop message + bias + relu)
// Safe in-place when out == acc.
// ---------------------------------------------------------------------------
template <int F>
__global__ void gcn_selfloop_bias_relu(const float* __restrict__ acc,
                                       const float* __restrict__ T,
                                       const float* __restrict__ dinv,
                                       const float* __restrict__ b,
                                       float* __restrict__ out, int n) {
    int idx = blockIdx.x * blockDim.x + threadIdx.x;
    int i = idx / F;
    int f = idx - i * F;
    if (i >= n) return;
    float di = dinv[i];
    float v = acc[idx] + T[idx] * di * di + b[f];
    out[idx] = v > 0.0f ? v : 0.0f;
}

// ---------------------------------------------------------------------------
// Global mean pool (sum + counts via atomics), F2 = 128 features.
// ---------------------------------------------------------------------------
__global__ void gcn_pool(const float* __restrict__ h2, const int* __restrict__ batch,
                         float* __restrict__ pooled, float* __restrict__ counts, int n) {
    int idx = blockIdx.x * blockDim.x + threadIdx.x;
    int i = idx >> 7;
    int f = idx & 127;
    if (i >= n) return;
    int g = batch[i];
    atomicAdd(&pooled[(size_t)g * 128 + f], h2[idx]);
    if (f == 0) atomicAdd(&counts[g], 1.0f);
}

// ---------------------------------------------------------------------------
// FC head: out[g] = dot(pooled[g]/count[g], fcW) + fcb.   G=128, 2F=128.
// ---------------------------------------------------------------------------
__global__ void gcn_fc(const float* __restrict__ pooled, const float* __restrict__ counts,
                       const float* __restrict__ fcW, const float* __restrict__ fcb,
                       float* __restrict__ out, int g) {
    int gi = blockIdx.x * blockDim.x + threadIdx.x;
    if (gi >= g) return;
    float c = counts[gi];
    c = c > 1.0f ? c : 1.0f;
    float s = 0.0f;
    for (int f = 0; f < 128; ++f) s += pooled[(size_t)gi * 128 + f] * fcW[f];
    out[gi] = s / c + fcb[0];
}

extern "C" void kernel_launch(void* const* d_in, const int* in_sizes, int n_in,
                              void* d_out, int out_size, void* d_ws, size_t ws_size,
                              hipStream_t stream) {
    const float* x    = (const float*)d_in[0];
    const int*   ei   = (const int*)d_in[1];
    const int*   batch= (const int*)d_in[2];
    const float* W1   = (const float*)d_in[3];
    const float* b1   = (const float*)d_in[4];
    const float* W2   = (const float*)d_in[5];
    const float* b2   = (const float*)d_in[6];
    const float* fcW  = (const float*)d_in[7];
    const float* fcb  = (const float*)d_in[8];
    float*       out  = (float*)d_out;

    const int n = in_sizes[0] / 64;   // nodes
    const int e = in_sizes[1] / 2;    // edges
    const int g = out_size;           // graphs (NOUT=1)
    const int* src = ei;
    const int* dst = ei + e;

    // ---- workspace layout ----
    char* ws = (char*)d_ws;
    size_t off = 0;
    auto take = [&](size_t bytes) -> void* {
        void* p = ws + off;
        off += (bytes + 255) & ~(size_t)255;
        return p;
    };
    float* deg    = (float*)take((size_t)n * 4);
    float* dinv   = (float*)take((size_t)n * 4);
    float* T1     = (float*)take((size_t)n * 64 * 4);
    float* acc1   = (float*)take((size_t)n * 64 * 4);   // becomes h1 in place
    float* T2     = (float*)take((size_t)n * 128 * 4);
    float* acc2   = (float*)take((size_t)n * 128 * 4);  // becomes h2 in place
    float* pooled = (float*)take((size_t)g * 128 * 4);
    float* counts = (float*)take((size_t)g * 4);
    (void)ws_size; (void)n_in;

    hipMemsetAsync(deg,    0, (size_t)n * 4,        stream);
    hipMemsetAsync(acc1,   0, (size_t)n * 64 * 4,   stream);
    hipMemsetAsync(acc2,   0, (size_t)n * 128 * 4,  stream);
    hipMemsetAsync(pooled, 0, (size_t)g * 128 * 4,  stream);
    hipMemsetAsync(counts, 0, (size_t)g * 4,        stream);

    const int BLK = 256;
    // degrees + normalization
    gcn_degree<<<(e + BLK - 1) / BLK, BLK, 0, stream>>>(dst, deg, e);
    gcn_dinv  <<<(n + BLK - 1) / BLK, BLK, 0, stream>>>(deg, dinv, n);

    const int ntiles = (n + 15) / 16;            // 3125 (exact)
    const int gblk   = (ntiles + 3) / 4;         // 4 waves/block

    // Layer 1: T1 = x @ W1 ; scatter ; h1 = relu(acc1 + selfloop + b1)
    gcn_gemm_wmma<64><<<gblk, 128, 0, stream>>>(x, W1, T1, ntiles);
    {
        long long tot = (long long)e * 64;
        gcn_scatter<64><<<(int)((tot + BLK - 1) / BLK), BLK, 0, stream>>>(T1, src, dst, dinv, acc1, e);
        long long nt = (long long)n * 64;
        gcn_selfloop_bias_relu<64><<<(int)((nt + BLK - 1) / BLK), BLK, 0, stream>>>(acc1, T1, dinv, b1, acc1, n);
    }

    // Layer 2: T2 = h1 @ W2 ; scatter ; h2 = relu(acc2 + selfloop + b2)
    gcn_gemm_wmma<128><<<gblk, 128, 0, stream>>>(acc1, W2, T2, ntiles);
    {
        long long tot = (long long)e * 128;
        gcn_scatter<128><<<(int)((tot + BLK - 1) / BLK), BLK, 0, stream>>>(T2, src, dst, dinv, acc2, e);
        long long nt = (long long)n * 128;
        gcn_selfloop_bias_relu<128><<<(int)((nt + BLK - 1) / BLK), BLK, 0, stream>>>(acc2, T2, dinv, b2, acc2, n);
    }

    // Pool + FC head
    {
        long long nt = (long long)n * 128;
        gcn_pool<<<(int)((nt + BLK - 1) / BLK), BLK, 0, stream>>>(acc2, batch, pooled, counts, n);
    }
    gcn_fc<<<(g + 127) / 128, 128, 0, stream>>>(pooled, counts, fcW, fcb, out, g);
}
